// Attention_11287174054323
// MI455X (gfx1250) — matrix operand order repaired and checked by reading, not compiled, not run on hardware
//
#include <hip/hip_runtime.h>

#define DIMW   2048
#define NHEADS 16
#define HDIM   128
#define BATCH  2
#define SEQ    2048
#define BS     (BATCH*SEQ)   // 4096
#define GK     2048          // K dim of all GEMMs

typedef __bf16 bf16_t;
typedef bf16_t v16bf __attribute__((ext_vector_type(16)));
typedef bf16_t v8bf  __attribute__((ext_vector_type(8)));
typedef float  v8f   __attribute__((ext_vector_type(8)));
typedef int    v4i   __attribute__((ext_vector_type(4)));

#if __has_builtin(__builtin_amdgcn_global_load_async_to_lds_b128)
#define ASYNC_LDS 1
#else
#define ASYNC_LDS 0
#endif

static __device__ __forceinline__ void async_cp16(const bf16_t* g, bf16_t* l) {
#if ASYNC_LDS
    __builtin_amdgcn_global_load_async_to_lds_b128(
        (__attribute__((address_space(1))) v4i*)g,
        (__attribute__((address_space(3))) v4i*)l, 0, 0);
#endif
}

static __device__ __forceinline__ void wait_async() {
#if ASYNC_LDS
 #if __has_builtin(__builtin_amdgcn_s_wait_asynccnt)
    __builtin_amdgcn_s_wait_asynccnt(0);
 #else
    asm volatile("s_wait_asynccnt 0" ::: "memory");
 #endif
#endif
}

static __device__ __forceinline__ v8f vzero8f() {
    v8f r = {0.f,0.f,0.f,0.f,0.f,0.f,0.f,0.f};
    return r;
}

static __device__ __forceinline__ v16bf cat8(v8bf lo, v8bf hi) {
    v16bf r;
#pragma unroll
    for (int i = 0; i < 8; ++i) { r[i] = lo[i]; r[i + 8] = hi[i]; }
    return r;
}

// D = A(16x32) * B(32x16) + C, bf16 inputs, f32 accumulate
static __device__ __forceinline__ v8f wmma_bf16(v16bf a, v16bf b, v8f c) {
    return __builtin_amdgcn_wmma_f32_16x16x32_bf16(
        false, a, false, b, (short)0, c, false, false);
}

static __device__ __forceinline__ void lds_fence() {
    asm volatile("s_wait_dscnt 0" ::: "memory");
}

// ---------------------------------------------------------------------------
// fp32 -> bf16 elementwise convert
// ---------------------------------------------------------------------------
__global__ void k_f32_to_bf16(const float* __restrict__ src,
                              bf16_t* __restrict__ dst, int n) {
    int i = blockIdx.x * blockDim.x + threadIdx.x;
    int stride = gridDim.x * blockDim.x;
    for (; i < n; i += stride) dst[i] = (bf16_t)src[i];
}

// ---------------------------------------------------------------------------
// fp32 [R,C] -> bf16 transposed [C,R]
// ---------------------------------------------------------------------------
__global__ void k_transpose_bf16(const float* __restrict__ src,
                                 bf16_t* __restrict__ dst, int R, int C) {
    __shared__ float tile[32][33];
    int c0 = blockIdx.x * 32, r0 = blockIdx.y * 32;
#pragma unroll
    for (int j = 0; j < 4; ++j)
        tile[threadIdx.y + j * 8][threadIdx.x] =
            src[(size_t)(r0 + threadIdx.y + j * 8) * C + c0 + threadIdx.x];
    __syncthreads();
#pragma unroll
    for (int j = 0; j < 4; ++j)
        dst[(size_t)(c0 + threadIdx.y + j * 8) * R + r0 + threadIdx.x] =
            (bf16_t)tile[threadIdx.x][threadIdx.y + j * 8];
}

// ---------------------------------------------------------------------------
// GEMM: C[M,N] = A[M,GK] * W, BT = W^T stored [N,GK], all bf16.
// Block = 256 threads (8 waves), 128x128 tile, K staged 64-wide, double-
// buffered LDS (async-to-LDS when available). Wave = 64x32 (4x2 WMMA tiles).
// MODE 0: bf16 row-major. MODE 1: bf16 scattered as V^T[b,h,d,s]. MODE 2: f32.
// ---------------------------------------------------------------------------
template <int MODE>
__global__ __launch_bounds__(256)
void k_gemm(const bf16_t* __restrict__ A, const bf16_t* __restrict__ BT,
            void* __restrict__ Cout, int N) {
    // 128 rows x 64 elems, padded to 72 (144B stride = 36 banks: conflict-free)
    __shared__ bf16_t ldsA[2][128 * 72];
    __shared__ bf16_t ldsB[2][128 * 72];

    const int t    = threadIdx.x;
    const int w    = t >> 5;
    const int lane = t & 31;
    const int half = lane >> 4;
    const int n16  = lane & 15;
    const int wm   = w >> 2;  // 0..1
    const int wn   = w & 3;   // 0..3
    const int m0   = blockIdx.y * 128;
    const int n0   = blockIdx.x * 128;

    const int rowf = t >> 1;  // 0..127
    const int seg  = t & 1;   // 0..1 (each thread stages 64B per matrix)

    v8f acc[4][2];
#pragma unroll
    for (int i = 0; i < 4; ++i)
#pragma unroll
        for (int j = 0; j < 2; ++j) acc[i][j] = vzero8f();

    auto stage = [&](int buf, int kk) {
        const bf16_t* ga = A  + (size_t)(m0 + rowf) * GK + kk + seg * 32;
        const bf16_t* gb = BT + (size_t)(n0 + rowf) * GK + kk + seg * 32;
        bf16_t* la = &ldsA[buf][rowf * 72 + seg * 32];
        bf16_t* lb = &ldsB[buf][rowf * 72 + seg * 32];
#if ASYNC_LDS
#pragma unroll
        for (int j = 0; j < 4; ++j) {
            async_cp16(ga + j * 8, la + j * 8);
            async_cp16(gb + j * 8, lb + j * 8);
        }
#else
#pragma unroll
        for (int j = 0; j < 4; ++j) {
            v8bf av = *(const v8bf*)(ga + j * 8);
            v8bf bv = *(const v8bf*)(gb + j * 8);
            *(v8bf*)(la + j * 8) = av;
            *(v8bf*)(lb + j * 8) = bv;
        }
#endif
    };

    auto compute = [&](int buf) {
#pragma unroll
        for (int k2 = 0; k2 < 2; ++k2) {
            v16bf af[4], bfr[2];
#pragma unroll
            for (int ms = 0; ms < 4; ++ms) {
                int row = wm * 64 + ms * 16 + n16;
                const bf16_t* p = &ldsA[buf][row * 72 + k2 * 32];
                af[ms] = cat8(*(const v8bf*)(p + half * 8),
                              *(const v8bf*)(p + 16 + half * 8));
            }
#pragma unroll
            for (int ns = 0; ns < 2; ++ns) {
                int row = wn * 32 + ns * 16 + n16;
                const bf16_t* p = &ldsB[buf][row * 72 + k2 * 32];
                bfr[ns] = cat8(*(const v8bf*)(p + half * 8),
                               *(const v8bf*)(p + 16 + half * 8));
            }
#pragma unroll
            for (int ms = 0; ms < 4; ++ms)
#pragma unroll
                for (int ns = 0; ns < 2; ++ns)
                    acc[ms][ns] = wmma_bf16(af[ms], bfr[ns], acc[ms][ns]);
        }
    };

    stage(0, 0);
    wait_async();
    __syncthreads();
    int cur = 0;
#pragma unroll 1
    for (int kk = 0; kk < GK; kk += 64) {
        if (kk + 64 < GK) stage(cur ^ 1, kk + 64);
        compute(cur);
        wait_async();
        __syncthreads();
        cur ^= 1;
    }

    // epilogue: C layout -> lane col n = n16, rows m = half*8 + r
#pragma unroll
    for (int ms = 0; ms < 4; ++ms)
#pragma unroll
        for (int ns = 0; ns < 2; ++ns) {
            int m_base = m0 + wm * 64 + ms * 16 + half * 8;
            int n      = n0 + wn * 32 + ns * 16 + n16;
#pragma unroll
            for (int r = 0; r < 8; ++r) {
                int m   = m_base + r;
                float v = acc[ms][ns][r];
                if (MODE == 0) {
                    ((bf16_t*)Cout)[(size_t)m * N + n] = (bf16_t)v;
                } else if (MODE == 2) {
                    ((float*)Cout)[(size_t)m * N + n] = v;
                } else {  // V^T[b, h, d, s]
                    int b = m >> 11, s = m & (SEQ - 1);
                    int h = n >> 7, d = n & (HDIM - 1);
                    ((bf16_t*)Cout)[(size_t)((b * NHEADS + h) * HDIM + d) * SEQ + s] =
                        (bf16_t)v;
                }
            }
        }
}

// ---------------------------------------------------------------------------
// RoPE in place on Q and K (bf16 [B*S, H*D]); folds 1/sqrt(D) into Q.
// ---------------------------------------------------------------------------
__global__ void k_rope(bf16_t* __restrict__ Q, bf16_t* __restrict__ Kc) {
    int idx = blockIdx.x * blockDim.x + threadIdx.x;  // B*S*H*64
    int i = idx & 63;
    int h = (idx >> 6) & (NHEADS - 1);
    int s = (idx >> 10) & (SEQ - 1);
    int b = idx >> 21;
    float invf = __expf(-(float)i * (9.210340371976184f / 64.0f));
    float ang  = (float)s * invf;
    float sn, c;
    __sincosf(ang, &sn, &c);
    size_t base = ((size_t)(b * SEQ + s)) * DIMW + h * HDIM + i;
    const float scale = 0.08838834764831845f;  // 1/sqrt(128)
    {
        float q0 = (float)Q[base], q1 = (float)Q[base + 64];
        Q[base]      = (bf16_t)((q0 * c - q1 * sn) * scale);
        Q[base + 64] = (bf16_t)((q1 * c + q0 * sn) * scale);
    }
    {
        float k0 = (float)Kc[base], k1 = (float)Kc[base + 64];
        Kc[base]      = (bf16_t)(k0 * c - k1 * sn);
        Kc[base + 64] = (bf16_t)(k1 * c + k0 * sn);
    }
}

// ---------------------------------------------------------------------------
// Flash attention. Grid: (B*H, SEQ/128); block 256 = 8 waves.
// Each wave owns 16 query rows. K/V for each 32-key chunk are staged once
// per block into double-buffered LDS (async-to-LDS when available) and
// shared by all 8 waves.
// ---------------------------------------------------------------------------
__global__ __launch_bounds__(256)
void k_attn(const bf16_t* __restrict__ Q, const bf16_t* __restrict__ Kc,
            const bf16_t* __restrict__ Vt, bf16_t* __restrict__ O) {
    __shared__ bf16_t ldsK[2][32 * 136];  // 32 keys x 128 d, pad 128->136
    __shared__ bf16_t ldsV[2][128 * 40];  // 128 d x 32 keys, pad 32->40
    __shared__ bf16_t ldsP[8][16 * 40];   // per-wave P tile

    const int t    = threadIdx.x;
    const int w    = t >> 5;
    const int lane = t & 31;
    const int half = lane >> 4;
    const int n16  = lane & 15;
    const int bh   = blockIdx.x;
    const int b    = bh >> 4, h = bh & (NHEADS - 1);
    const int q0   = blockIdx.y * 128 + w * 16;

    const int rowK = t >> 3, segK = t & 7;  // K slab: 32 rows x 256B
    const int rowV = t >> 1, segV = t & 1;  // V slab: 128 rows x 64B

    auto stage = [&](int buf, int kk) {
        const bf16_t* gk = Kc + (size_t)(b * SEQ + kk + rowK) * DIMW + h * HDIM + segK * 16;
        const bf16_t* gv = Vt + (size_t)(bh * HDIM + rowV) * SEQ + kk + segV * 16;
        bf16_t* lk = &ldsK[buf][rowK * 136 + segK * 16];
        bf16_t* lv = &ldsV[buf][rowV * 40 + segV * 16];
#if ASYNC_LDS
        async_cp16(gk, lk);
        async_cp16(gk + 8, lk + 8);
        async_cp16(gv, lv);
        async_cp16(gv + 8, lv + 8);
#else
        v8bf k0 = *(const v8bf*)gk;
        v8bf k1 = *(const v8bf*)(gk + 8);
        v8bf v0 = *(const v8bf*)gv;
        v8bf v1 = *(const v8bf*)(gv + 8);
        *(v8bf*)lk = k0; *(v8bf*)(lk + 8) = k1;
        *(v8bf*)lv = v0; *(v8bf*)(lv + 8) = v1;
#endif
    };

    // Q fragments (A-frag, kept in registers for whole key sweep)
    v16bf qf[4];
    {
        const bf16_t* qp = Q + ((size_t)(b * SEQ + q0 + n16)) * DIMW + h * HDIM;
#pragma unroll
        for (int c = 0; c < 4; ++c) {
            v8bf lo = *(const v8bf*)(qp + c * 32 + half * 16);
            v8bf hi = *(const v8bf*)(qp + c * 32 + half * 16 + 8);
            qf[c]   = cat8(lo, hi);
        }
    }

    v8f o[8];
#pragma unroll
    for (int i = 0; i < 8; ++i) o[i] = vzero8f();
    float mrow[8], lrow[8];
#pragma unroll
    for (int r = 0; r < 8; ++r) { mrow[r] = -1e30f; lrow[r] = 0.0f; }

    bf16_t* pbase = &ldsP[w][0];

    stage(0, 0);
    wait_async();
    __syncthreads();
    int cur = 0;

#pragma unroll 1
    for (int kk = 0; kk < SEQ; kk += 32) {
        if (kk + 32 < SEQ) stage(cur ^ 1, kk + 32);

        // scores: 2 key sub-tiles x 4 d-chunks, K B-frags from LDS slab
        v8f s0 = vzero8f(), s1 = vzero8f();
#pragma unroll
        for (int c = 0; c < 4; ++c) {
            const bf16_t* kp0 = &ldsK[cur][(n16)*136 + c * 32 + half * 16];
            v16bf kf0 = cat8(*(const v8bf*)kp0, *(const v8bf*)(kp0 + 8));
            s0 = wmma_bf16(qf[c], kf0, s0);
            const bf16_t* kp1 = kp0 + 16 * 136;
            v16bf kf1 = cat8(*(const v8bf*)kp1, *(const v8bf*)(kp1 + 8));
            s1 = wmma_bf16(qf[c], kf1, s1);
        }

        // online softmax (row = query = half*8 + r; keys spread over 16 lanes)
        float p0[8], p1[8];
#pragma unroll
        for (int r = 0; r < 8; ++r) {
            float mx = fmaxf(s0[r], s1[r]);
            mx = fmaxf(mx, __shfl_xor(mx, 1));
            mx = fmaxf(mx, __shfl_xor(mx, 2));
            mx = fmaxf(mx, __shfl_xor(mx, 4));
            mx = fmaxf(mx, __shfl_xor(mx, 8));
            float mnew = fmaxf(mrow[r], mx);
            float corr = __expf(mrow[r] - mnew);
            float e0 = __expf(s0[r] - mnew);
            float e1 = __expf(s1[r] - mnew);
            float sum = e0 + e1;
            sum += __shfl_xor(sum, 1);
            sum += __shfl_xor(sum, 2);
            sum += __shfl_xor(sum, 4);
            sum += __shfl_xor(sum, 8);
            lrow[r] = lrow[r] * corr + sum;
            mrow[r] = mnew;
            p0[r] = e0;
            p1[r] = e1;
#pragma unroll
            for (int d8 = 0; d8 < 8; ++d8) o[d8][r] *= corr;
        }

        // P: C-layout -> A-fragment via per-wave LDS round trip
#pragma unroll
        for (int r = 0; r < 8; ++r) {
            pbase[(half * 8 + r) * 40 + n16]      = (bf16_t)p0[r];
            pbase[(half * 8 + r) * 40 + 16 + n16] = (bf16_t)p1[r];
        }
        lds_fence();
        v16bf pf;
        {
            v8bf lo = *(const v8bf*)&pbase[n16 * 40 + half * 16];
            v8bf hi = *(const v8bf*)&pbase[n16 * 40 + half * 16 + 8];
            pf = cat8(lo, hi);
        }

        // O += P * V : V B-frags from LDS slab (contiguous keys per d-row)
#pragma unroll
        for (int d8 = 0; d8 < 8; ++d8) {
            const bf16_t* vp = &ldsV[cur][(d8 * 16 + n16) * 40 + half * 16];
            v16bf vf = cat8(*(const v8bf*)vp, *(const v8bf*)(vp + 8));
            o[d8] = wmma_bf16(pf, vf, o[d8]);
        }

        wait_async();
        __syncthreads();
        cur ^= 1;
    }

#pragma unroll
    for (int r = 0; r < 8; ++r) lrow[r] = 1.0f / lrow[r];
#pragma unroll
    for (int d8 = 0; d8 < 8; ++d8)
#pragma unroll
        for (int r = 0; r < 8; ++r) {
            int srow = q0 + half * 8 + r;
            O[((size_t)(b * SEQ + srow)) * DIMW + h * HDIM + d8 * 16 + n16] =
                (bf16_t)(o[d8][r] * lrow[r]);
        }
}

// ---------------------------------------------------------------------------
extern "C" void kernel_launch(void* const* d_in, const int* in_sizes, int n_in,
                              void* d_out, int out_size, void* d_ws, size_t ws_size,
                              hipStream_t stream) {
    (void)in_sizes; (void)n_in; (void)out_size; (void)ws_size;
    const float* x  = (const float*)d_in[0];
    const float* Wq = (const float*)d_in[1];
    const float* Wk = (const float*)d_in[2];
    const float* Wv = (const float*)d_in[3];
    const float* Wo = (const float*)d_in[4];

    char* ws = (char*)d_ws;
    const size_t MB = 1024ull * 1024ull;
    bf16_t* xb  = (bf16_t*)(ws);             // 16 MB  (reused as Ob after V GEMM)
    bf16_t* WqT = (bf16_t*)(ws + 16 * MB);   //  8 MB
    bf16_t* WkT = (bf16_t*)(ws + 24 * MB);
    bf16_t* WvT = (bf16_t*)(ws + 32 * MB);
    bf16_t* WoT = (bf16_t*)(ws + 40 * MB);
    bf16_t* Qb  = (bf16_t*)(ws + 48 * MB);   // 16 MB
    bf16_t* Kb  = (bf16_t*)(ws + 64 * MB);   // 16 MB
    bf16_t* Vt  = (bf16_t*)(ws + 80 * MB);   // 16 MB -> total 96 MB
    bf16_t* Ob  = xb;                        // alias: x_bf16 dead after V GEMM

    k_f32_to_bf16<<<4096, 256, 0, stream>>>(x, xb, BS * DIMW);

    dim3 tb(32, 8), tg(DIMW / 32, DIMW / 32);
    k_transpose_bf16<<<tg, tb, 0, stream>>>(Wq, WqT, DIMW, DIMW);
    k_transpose_bf16<<<tg, tb, 0, stream>>>(Wk, WkT, DIMW, DIMW);
    k_transpose_bf16<<<tg, tb, 0, stream>>>(Wv, WvT, DIMW, DIMW);
    k_transpose_bf16<<<tg, tb, 0, stream>>>(Wo, WoT, DIMW, DIMW);

    dim3 gg(DIMW / 128, BS / 128);  // (16, 32)
    k_gemm<0><<<gg, 256, 0, stream>>>(xb, WqT, Qb, DIMW);
    k_gemm<0><<<gg, 256, 0, stream>>>(xb, WkT, Kb, DIMW);
    k_gemm<1><<<gg, 256, 0, stream>>>(xb, WvT, Vt, DIMW);

    k_rope<<<(BS * NHEADS * 64) / 256, 256, 0, stream>>>(Qb, Kb);

    k_attn<<<dim3(BATCH * NHEADS, SEQ / 128), 256, 0, stream>>>(Qb, Kb, Vt, Ob);

    k_gemm<2><<<gg, 256, 0, stream>>>(Ob, WoT, (float*)d_out, DIMW);
}
